// LSTMModel_15040975470762
// MI455X (gfx1250) — compile-verified, matrix operand order checked by reading, not run on hardware
//
#include <hip/hip_runtime.h>

// ---------------------------------------------------------------------------
// Fused 2-layer LSTM (H=50, T=512) + FC head for MI455X (gfx1250, wave32).
// One workgroup (128 threads = 4 waves) owns a 16-row batch tile end-to-end.
// Recurrent GEMMs run on v_wmma_f32_16x16x32_f16; weights + h-state in LDS.
// Wave id is readfirstlane'd so WMMA tile loops are scalar (EXEC stays all-1s).
// ---------------------------------------------------------------------------

typedef __attribute__((ext_vector_type(16))) _Float16 v16h;
typedef __attribute__((ext_vector_type(8)))  _Float16 v8h;
typedef __attribute__((ext_vector_type(8)))  float    v8f;

#define TSTEPS 512
#define HDIM   50
#define GATES  200         // 4*H
#define NTILES 13          // ceil(200/16)
#define NCOLS  (NTILES*16) // 208 padded gate columns
#define KPAD   64          // H padded to 2 x K32 WMMA tiles
#define MTILE  16          // batch rows per workgroup
#define NTHREADS 128

// LDS layout (bytes):
//   gbuf  [NCOLS][16] f32        = 13312
//   hlast [16][KPAD]  f32        =  4096
//   xbuf  [16]        f32        =    64
//   W1    [NCOLS][KPAD] f16      = 26624   (Whh1, per-column K-major, zero pad)
//   W2i   [NCOLS][KPAD] f16      = 26624   (Wih2)
//   W2h   [NCOLS][KPAD] f16      = 26624   (Whh2)
//   h1b   [16][KPAD]  f16        =  2048
//   h2b   [16][KPAD]  f16        =  2048
static constexpr int SMEM_BYTES =
    NCOLS*16*4 + MTILE*KPAD*4 + 16*4 + 3*NCOLS*KPAD*2 + 2*MTILE*KPAD*2; // 101440

__device__ __forceinline__ float sigm(float x) {
  return 1.0f / (1.0f + __expf(-x));
}
__device__ __forceinline__ float tanh_fast(float x) {
  float e = __expf(2.0f * x);
  return (e - 1.0f) / (e + 1.0f);
}

// Load one 16x32 f16 WMMA fragment (A or B role) from an LDS tile laid out
// [16 rows/cols][KPAD halves]. Per ISA 7.12.2: lane L holds M=L%16, with K
// chunks {kb..kb+7, kb+16..kb+23}, kb = 32*ktile + (L>=16 ? 8 : 0).
__device__ __forceinline__ v16h ldfrag(const _Float16* base, int ktile) {
  const int lane = threadIdx.x & 31;
  const int r    = lane & 15;
  const int kb   = ktile * 32 + ((lane & 16) ? 8 : 0);
  const _Float16* p = base + r * KPAD + kb;
  v8h lo = *(const v8h*)(p);
  v8h hi = *(const v8h*)(p + 16);
  return __builtin_shufflevector(lo, hi, 0,1,2,3,4,5,6,7,8,9,10,11,12,13,14,15);
}

__device__ __forceinline__ v8f wmma16(v16h a, v16h b, v8f c) {
  // D = A*B + C, f16 inputs, f32 accumulate
  return __builtin_amdgcn_wmma_f32_16x16x32_f16(
      /*neg_a=*/false, a, /*neg_b=*/false, b,
      /*c_mod=*/(short)0, c, /*reuse_a=*/false, /*reuse_b=*/false);
}

// Scatter a 16x16 f32 accumulator tile to gbuf laid out [col][16 rows].
// C layout: VGPR i holds row (i or i+8), col = lane%16 -> rows 0..7 contiguous.
__device__ __forceinline__ void st_acc(float* gbuf, int c0, v8f a) {
  const int lane = threadIdx.x & 31;
  float* p = gbuf + (c0 + (lane & 15)) * 16 + ((lane & 16) ? 8 : 0);
  *(float4*)(p)     = make_float4(a[0], a[1], a[2], a[3]);
  *(float4*)(p + 4) = make_float4(a[4], a[5], a[6], a[7]);
}

__global__ void __launch_bounds__(NTHREADS)
lstm_fused_kernel(const float* __restrict__ x,
                  const float* __restrict__ Wih1, const float* __restrict__ Whh1,
                  const float* __restrict__ bih1, const float* __restrict__ bhh1,
                  const float* __restrict__ Wih2, const float* __restrict__ Whh2,
                  const float* __restrict__ bih2, const float* __restrict__ bhh2,
                  const float* __restrict__ Wfc1, const float* __restrict__ bfc1,
                  const float* __restrict__ Wfc2, const float* __restrict__ bfc2,
                  float* __restrict__ out) {
  extern __shared__ char smem[];
  float*    gbuf  = (float*)smem;                 // [NCOLS][16]
  float*    hlast = gbuf + NCOLS * 16;            // [16][KPAD]
  float*    xbuf  = hlast + MTILE * KPAD;         // [16]
  _Float16* W1    = (_Float16*)(xbuf + 16);       // [NCOLS][KPAD]
  _Float16* W2i   = W1  + NCOLS * KPAD;
  _Float16* W2h   = W2i + NCOLS * KPAD;
  _Float16* h1b   = W2h + NCOLS * KPAD;           // [16][KPAD]
  _Float16* h2b   = h1b + MTILE * KPAD;

  const int tid = threadIdx.x;
  const int b0  = blockIdx.x * MTILE;

  // ---- one-time weight staging into LDS (f16, zero padded) ----
  for (int idx = tid; idx < NCOLS * KPAD; idx += NTHREADS) {
    const int col = idx >> 6;      // padded gate column (== Whh row for col<200)
    const int j   = idx & 63;      // hidden input index (K)
    const bool v  = (col < GATES) && (j < HDIM);
    W1 [idx] = v ? (_Float16)Whh1[col * HDIM + j] : (_Float16)0.0f;
    W2i[idx] = v ? (_Float16)Wih2[col * HDIM + j] : (_Float16)0.0f;
    W2h[idx] = v ? (_Float16)Whh2[col * HDIM + j] : (_Float16)0.0f;
  }
  for (int idx = tid; idx < MTILE * KPAD; idx += NTHREADS) {
    h1b[idx] = (_Float16)0.0f;
    h2b[idx] = (_Float16)0.0f;
  }

  // ---- per-thread elementwise ownership: (row, k = kb..kb+7) ----
  const int row = tid & 15;
  const int kb  = (tid >> 4) * 8;
  float wc[8][4], b1c[8][4], b2c[8][4], c1[8], c2[8];
#pragma unroll
  for (int i = 0; i < 8; ++i) {
    const int k = kb + i;
    const bool v = (k < HDIM);
#pragma unroll
    for (int g = 0; g < 4; ++g) {
      const int w = g * HDIM + (v ? k : 0);
      wc [i][g] = v ? Wih1[w]             : 0.0f;   // Wih1 is (200,1)
      b1c[i][g] = v ? (bih1[w] + bhh1[w]) : 0.0f;
      b2c[i][g] = v ? (bih2[w] + bhh2[w]) : 0.0f;
    }
    c1[i] = 0.0f;
    c2[i] = 0.0f;
  }

  // N-tile split across the 4 waves: {4,4,3,2} of 13 tiles.
  // readfirstlane pins the wave id into an SGPR so the tile loops below get
  // scalar control flow (no exec-mask loop; EXEC stays all-1s for WMMA).
  const int wv = __builtin_amdgcn_readfirstlane(tid >> 5);
  const int nts[5] = {0, 4, 8, 11, 13};
  const int nt_lo = nts[wv], nt_hi = nts[wv + 1];

  __syncthreads();

  for (int t = 0; t < TSTEPS; ++t) {
    if (tid < 16) {
      const float* xp = x + (size_t)(b0 + tid) * TSTEPS + t;
      xbuf[tid] = *xp;
      __builtin_prefetch(xp + 1, 0, 1);   // global_prefetch for step t+1
    }

    // ---- phase A: g1 = h1 @ Whh1^T (layer-1 recurrent gates) ----
    {
      v16h a0 = ldfrag(h1b, 0), a1 = ldfrag(h1b, 1);
      for (int nt = nt_lo; nt < nt_hi; ++nt) {
        const int c0 = nt * 16;
        v8f acc = {};
        acc = wmma16(a0, ldfrag(W1 + c0 * KPAD, 0), acc);
        acc = wmma16(a1, ldfrag(W1 + c0 * KPAD, 1), acc);
        st_acc(gbuf, c0, acc);
      }
    }
    __syncthreads();

    // ---- phase B: layer-1 cell update (input proj folded in: x*Wih1) ----
    {
      const float xv = xbuf[row];
#pragma unroll
      for (int i = 0; i < 8; ++i) {
        const int k = kb + i;
        if (k < HDIM) {
          float gi = gbuf[(0 * HDIM + k) * 16 + row] + xv * wc[i][0] + b1c[i][0];
          float gf = gbuf[(1 * HDIM + k) * 16 + row] + xv * wc[i][1] + b1c[i][1];
          float gg = gbuf[(2 * HDIM + k) * 16 + row] + xv * wc[i][2] + b1c[i][2];
          float go = gbuf[(3 * HDIM + k) * 16 + row] + xv * wc[i][3] + b1c[i][3];
          float iv = sigm(gi), fv = sigm(gf), gv = tanh_fast(gg), ov = sigm(go);
          c1[i] = fv * c1[i] + iv * gv;
          float h = ov * tanh_fast(c1[i]);
          h1b[row * KPAD + k] = (_Float16)h;
        }
      }
    }
    __syncthreads();

    // ---- phase C: g2 = h1_t @ Wih2^T + h2 @ Whh2^T ----
    {
      v16h a10 = ldfrag(h1b, 0), a11 = ldfrag(h1b, 1);
      v16h a20 = ldfrag(h2b, 0), a21 = ldfrag(h2b, 1);
      for (int nt = nt_lo; nt < nt_hi; ++nt) {
        const int c0 = nt * 16;
        v8f acc = {};
        acc = wmma16(a10, ldfrag(W2i + c0 * KPAD, 0), acc);
        acc = wmma16(a11, ldfrag(W2i + c0 * KPAD, 1), acc);
        acc = wmma16(a20, ldfrag(W2h + c0 * KPAD, 0), acc);
        acc = wmma16(a21, ldfrag(W2h + c0 * KPAD, 1), acc);
        st_acc(gbuf, c0, acc);
      }
    }
    __syncthreads();

    // ---- phase D: layer-2 cell update ----
    {
#pragma unroll
      for (int i = 0; i < 8; ++i) {
        const int k = kb + i;
        if (k < HDIM) {
          float gi = gbuf[(0 * HDIM + k) * 16 + row] + b2c[i][0];
          float gf = gbuf[(1 * HDIM + k) * 16 + row] + b2c[i][1];
          float gg = gbuf[(2 * HDIM + k) * 16 + row] + b2c[i][2];
          float go = gbuf[(3 * HDIM + k) * 16 + row] + b2c[i][3];
          float iv = sigm(gi), fv = sigm(gf), gv = tanh_fast(gg), ov = sigm(go);
          c2[i] = fv * c2[i] + iv * gv;
          float h = ov * tanh_fast(c2[i]);
          h2b[row * KPAD + k]   = (_Float16)h;
          hlast[row * KPAD + k] = h;           // f32 copy for the FC head
        }
      }
    }
    __syncthreads();
  }

  // ---- FC head: (16,50) -> (16,25) -> (16,5); 80 threads, scalar ----
  if (tid < MTILE * 5) {
    const int r = tid / 5, o = tid % 5;
    float acc = bfc2[o];
    for (int j = 0; j < 25; ++j) {
      float t1 = bfc1[j];
#pragma unroll 10
      for (int k = 0; k < HDIM; ++k) t1 += hlast[r * KPAD + k] * Wfc1[j * HDIM + k];
      acc += t1 * Wfc2[o * 25 + j];
    }
    out[(size_t)(b0 + r) * 5 + o] = acc;
  }
}

extern "C" void kernel_launch(void* const* d_in, const int* in_sizes, int n_in,
                              void* d_out, int out_size, void* d_ws, size_t ws_size,
                              hipStream_t stream) {
  (void)n_in; (void)d_ws; (void)ws_size; (void)out_size;
  const float* x    = (const float*)d_in[0];
  const float* Wih1 = (const float*)d_in[1];
  const float* Whh1 = (const float*)d_in[2];
  const float* bih1 = (const float*)d_in[3];
  const float* bhh1 = (const float*)d_in[4];
  const float* Wih2 = (const float*)d_in[5];
  const float* Whh2 = (const float*)d_in[6];
  const float* bih2 = (const float*)d_in[7];
  const float* bhh2 = (const float*)d_in[8];
  const float* Wfc1 = (const float*)d_in[9];
  const float* bfc1 = (const float*)d_in[10];
  const float* Wfc2 = (const float*)d_in[11];
  const float* bfc2 = (const float*)d_in[12];
  float* out = (float*)d_out;

  const int B   = in_sizes[0] / TSTEPS;   // x is (B, T, 1)
  const int nwg = B / MTILE;              // 256 workgroups for B=4096

  // dynamic LDS ~99 KB: raise the per-kernel cap (idempotent, capture-safe)
  hipFuncSetAttribute((const void*)lstm_fused_kernel,
                      hipFuncAttributeMaxDynamicSharedMemorySize, SMEM_BYTES);

  lstm_fused_kernel<<<nwg, NTHREADS, SMEM_BYTES, stream>>>(
      x, Wih1, Whh1, bih1, bhh1, Wih2, Whh2, bih2, bhh2,
      Wfc1, bfc1, Wfc2, bfc2, out);
}
